// MultiHeadAttention_44787918962713
// MI455X (gfx1250) — matrix operand order repaired and checked by reading, not compiled, not run on hardware
//
#include <hip/hip_runtime.h>

#define B_SZ    2
#define S_LEN   2048
#define D_MODEL 1024
#define H_CNT   16
#define HD      64

typedef __attribute__((ext_vector_type(16))) __bf16 v16bf;
typedef __attribute__((ext_vector_type(8)))  float  v8f;

union Frag { v16bf v; uint4 q[2]; unsigned short s[16]; };
union Acc  { v8f   v; float f[8]; };

__device__ __forceinline__ unsigned short f32_to_bf16(float x) {
  union { float f; unsigned u; } c; c.f = x;
  unsigned r = c.u + 0x7fffu + ((c.u >> 16) & 1u);   // round-to-nearest-even
  return (unsigned short)(r >> 16);
}

// A fragment: 16(M) x 32(K) bf16, row-major source, K contiguous.
// lane holds row m = m0 + lane%16; element e<8 -> K = k0 + hi*8 + e,
// element e>=8 -> K = k0 + 16 + hi*8 + (e-8).  (CDNA5 ISA §7.12.2)
__device__ __forceinline__ v16bf load_a(const unsigned short* base, int ld, int m0, int k0) {
  int lane = threadIdx.x & 31;
  int m = m0 + (lane & 15), hi = lane >> 4;
  const unsigned short* p = base + (size_t)m * ld + k0 + hi * 8;
  Frag f;
  f.q[0] = *(const uint4*)p;
  f.q[1] = *(const uint4*)(p + 16);
  return f.v;
}

// B fragment: 32(K) x 16(N) bf16 where B[k,n] = src[n*ld + k].
// lane holds col n = n0 + lane%16; element e<8 -> K = k0 + hi*8 + e,
// element e>=8 -> K = k0 + 16 + hi*8 + (e-8).
__device__ __forceinline__ v16bf load_b(const unsigned short* base, int ld, int n0, int k0) {
  int lane = threadIdx.x & 31;
  int n = n0 + (lane & 15), hi = lane >> 4;
  const unsigned short* p = base + (size_t)n * ld + k0 + hi * 8;
  Frag f;
  f.q[0] = *(const uint4*)p;
  f.q[1] = *(const uint4*)(p + 16);
  return f.v;
}

__device__ __forceinline__ v8f wmma_bf16(v16bf a, v16bf b, v8f c) {
  return __builtin_amdgcn_wmma_f32_16x16x32_bf16(false, a, false, b, (short)0, c, false, false);
}

// ---------------- f32 -> bf16 conversion (8 elems/thread) ----------------
__global__ void cvt_bf16(const float* __restrict__ src, unsigned short* __restrict__ dst, int n8) {
  int i = blockIdx.x * blockDim.x + threadIdx.x;
  if (i >= n8) return;
  const float4* s = (const float4*)src + 2 * (size_t)i;
  float4 a = s[0], b = s[1];
  union { uint4 q; unsigned short h[8]; } o;
  o.h[0] = f32_to_bf16(a.x); o.h[1] = f32_to_bf16(a.y);
  o.h[2] = f32_to_bf16(a.z); o.h[3] = f32_to_bf16(a.w);
  o.h[4] = f32_to_bf16(b.x); o.h[5] = f32_to_bf16(b.y);
  o.h[6] = f32_to_bf16(b.z); o.h[7] = f32_to_bf16(b.w);
  ((uint4*)dst)[i] = o.q;
}

// ---------------- fused QKV projection ----------------
// One wave: 32(M) x 64(N) tile of X·Wᵀ; 8 WMMA per k-step, B reused 2x, A 4x.
// Q,K stored [b,h,s,hd]; V stored transposed [b,h,hd,s].
__global__ void __launch_bounds__(256) qkv_proj(
    const unsigned short* __restrict__ Xb,
    const unsigned short* __restrict__ Wq,
    const unsigned short* __restrict__ Wk,
    const unsigned short* __restrict__ Wv,
    unsigned short* __restrict__ Qb,
    unsigned short* __restrict__ Kb,
    unsigned short* __restrict__ Vt) {
  int wave  = (blockIdx.x * blockDim.x + threadIdx.x) >> 5;
  int nquad = wave & 15;           // 16 quads of 64 cols
  int mtile = (wave >> 4) & 127;   // 128 tiles of 32 rows
  int mat   = wave >> 11;          // 0=Q, 1=K, 2=V
  int m0 = mtile * 32, n0 = nquad * 64;
  const unsigned short* W = (mat == 0) ? Wq : (mat == 1) ? Wk : Wv;

  Acc acc[2][4];
#pragma unroll
  for (int i = 0; i < 2; ++i)
#pragma unroll
    for (int j = 0; j < 4; ++j)
#pragma unroll
      for (int r = 0; r < 8; ++r) acc[i][j].f[r] = 0.0f;

  v16bf a0 = load_a(Xb, D_MODEL, m0, 0);
  v16bf a1 = load_a(Xb, D_MODEL, m0 + 16, 0);
  for (int k0 = 0; k0 < D_MODEL; k0 += 32) {
    // issue all B fragments for this k-step up front
    v16bf b0 = load_b(W, D_MODEL, n0,      k0);
    v16bf b1 = load_b(W, D_MODEL, n0 + 16, k0);
    v16bf b2 = load_b(W, D_MODEL, n0 + 32, k0);
    v16bf b3 = load_b(W, D_MODEL, n0 + 48, k0);
    // prefetch next k-step's A fragments (redundant dummy load on last iter)
    int kn = (k0 + 32 < D_MODEL) ? k0 + 32 : 0;
    v16bf a0n = load_a(Xb, D_MODEL, m0, kn);
    v16bf a1n = load_a(Xb, D_MODEL, m0 + 16, kn);
    acc[0][0].v = wmma_bf16(a0, b0, acc[0][0].v);
    acc[1][0].v = wmma_bf16(a1, b0, acc[1][0].v);
    acc[0][1].v = wmma_bf16(a0, b1, acc[0][1].v);
    acc[1][1].v = wmma_bf16(a1, b1, acc[1][1].v);
    acc[0][2].v = wmma_bf16(a0, b2, acc[0][2].v);
    acc[1][2].v = wmma_bf16(a1, b2, acc[1][2].v);
    acc[0][3].v = wmma_bf16(a0, b3, acc[0][3].v);
    acc[1][3].v = wmma_bf16(a1, b3, acc[1][3].v);
    a0 = a0n; a1 = a1n;
  }

  int lane = threadIdx.x & 31, hi = lane >> 4, ln = lane & 15;
#pragma unroll
  for (int i = 0; i < 2; ++i)
#pragma unroll
    for (int j = 0; j < 4; ++j)
#pragma unroll
      for (int r = 0; r < 8; ++r) {
        int row = m0 + 16 * i + r + 8 * hi;   // 0..4095
        int col = n0 + 16 * j + ln;           // 0..1023
        int b   = row >> 11, s = row & (S_LEN - 1);
        int h   = col >> 6,  hd = col & (HD - 1);
        unsigned short v = f32_to_bf16(acc[i][j].f[r]);
        size_t bh = (size_t)(b * H_CNT + h);
        if (mat == 2)      Vt[(bh * HD + hd) * S_LEN + s] = v;
        else if (mat == 0) Qb[(bh * S_LEN + s) * HD + hd] = v;
        else               Kb[(bh * S_LEN + s) * HD + hd] = v;
      }
}

// ---------------- flash attention (one 16-row q-tile per wave) ----------------
__global__ void __launch_bounds__(32) attn(
    const unsigned short* __restrict__ Qb,
    const unsigned short* __restrict__ Kb,
    const unsigned short* __restrict__ Vt,
    const int* __restrict__ mask,
    unsigned short* __restrict__ Ab) {
  __shared__ __align__(16) unsigned short Pld[16 * 32];

  int q0 = blockIdx.x * 16;
  int bh = blockIdx.y;                       // 0..B*H-1
  int b  = bh >> 4, h = bh & 15;
  int lane = threadIdx.x & 31, hi = lane >> 4, ln = lane & 15;

  const unsigned short* Qh = Qb + (size_t)bh * S_LEN * HD;
  const unsigned short* Kh = Kb + (size_t)bh * S_LEN * HD;
  const unsigned short* Vh = Vt + (size_t)bh * HD * S_LEN;
  const int* Mb = mask + (size_t)b * S_LEN * S_LEN;

  v16bf aq0 = load_a(Qh, HD, q0, 0);
  v16bf aq1 = load_a(Qh, HD, q0, 32);

  Acc O[4];
#pragma unroll
  for (int j = 0; j < 4; ++j)
#pragma unroll
    for (int r = 0; r < 8; ++r) O[j].f[r] = 0.0f;

  float mrow[8], lrow[8];
#pragma unroll
  for (int r = 0; r < 8; ++r) { mrow[r] = -3.0e38f; lrow[r] = 0.0f; }

  // preload K fragments for kb = 0
  v16bf bk00 = load_b(Kh, HD, 0,  0);
  v16bf bk01 = load_b(Kh, HD, 0,  32);
  v16bf bk10 = load_b(Kh, HD, 16, 0);
  v16bf bk11 = load_b(Kh, HD, 16, 32);

  const float sc = 0.125f;                   // 1/sqrt(HD)
  for (int kb = 0; kb < S_LEN; kb += 32) {
    // V fragments + mask ints issued early; softmax VALU below hides them
    v16bf bv0 = load_b(Vh, S_LEN, 0,  kb);
    v16bf bv1 = load_b(Vh, S_LEN, 16, kb);
    v16bf bv2 = load_b(Vh, S_LEN, 32, kb);
    v16bf bv3 = load_b(Vh, S_LEN, 48, kb);
    int mv0[8], mv1[8];
#pragma unroll
    for (int r = 0; r < 8; ++r) {
      size_t rb = (size_t)(q0 + r + 8 * hi) * S_LEN + kb + ln;
      mv0[r] = Mb[rb];
      mv1[r] = Mb[rb + 16];
    }

    // scores with current K fragments
    Acc st[2];
    {
      v8f c = {};
      c = wmma_bf16(aq0, bk00, c);
      c = wmma_bf16(aq1, bk01, c);
      st[0].v = c;
      v8f d = {};
      d = wmma_bf16(aq0, bk10, d);
      d = wmma_bf16(aq1, bk11, d);
      st[1].v = d;
    }

    // prefetch next k-block's K fragments + mask lines
    int kbn = (kb + 32 < S_LEN) ? kb + 32 : 0;
    bk00 = load_b(Kh, HD, kbn,      0);
    bk01 = load_b(Kh, HD, kbn,      32);
    bk10 = load_b(Kh, HD, kbn + 16, 0);
    bk11 = load_b(Kh, HD, kbn + 16, 32);
    __builtin_prefetch(&Mb[(size_t)(q0 + ln) * S_LEN + kbn + hi * 16], 0, 0);

    // scale + mask (faithful: masked -> -1e-9)
#pragma unroll
    for (int r = 0; r < 8; ++r) {
      float s0 = st[0].f[r] * sc, s1 = st[1].f[r] * sc;
      st[0].f[r] = (mv0[r] == 0) ? -1e-9f : s0;
      st[1].f[r] = (mv1[r] == 0) ? -1e-9f : s1;
    }
    // online softmax: row r+8*hi lives across the 16 lanes of this half-wave
#pragma unroll
    for (int r = 0; r < 8; ++r) {
      float mx = fmaxf(st[0].f[r], st[1].f[r]);
      mx = fmaxf(mx, __shfl_xor(mx, 1));
      mx = fmaxf(mx, __shfl_xor(mx, 2));
      mx = fmaxf(mx, __shfl_xor(mx, 4));
      mx = fmaxf(mx, __shfl_xor(mx, 8));
      float mnew  = fmaxf(mrow[r], mx);
      float alpha = __expf(mrow[r] - mnew);
      mrow[r] = mnew;
      float p0 = __expf(st[0].f[r] - mnew);
      float p1 = __expf(st[1].f[r] - mnew);
      st[0].f[r] = p0; st[1].f[r] = p1;
      float sm = p0 + p1;
      sm += __shfl_xor(sm, 1);
      sm += __shfl_xor(sm, 2);
      sm += __shfl_xor(sm, 4);
      sm += __shfl_xor(sm, 8);
      lrow[r] = lrow[r] * alpha + sm;
#pragma unroll
      for (int j = 0; j < 4; ++j) O[j].f[r] *= alpha;
    }
    // C-layout -> A-layout via LDS (single-wave workgroup)
#pragma unroll
    for (int t = 0; t < 2; ++t)
#pragma unroll
      for (int r = 0; r < 8; ++r)
        Pld[(r + 8 * hi) * 32 + t * 16 + ln] = f32_to_bf16(st[t].f[r]);
    __syncthreads();
    v16bf ap = load_a(Pld, 32, 0, 0);
    O[0].v = wmma_bf16(ap, bv0, O[0].v);
    O[1].v = wmma_bf16(ap, bv1, O[1].v);
    O[2].v = wmma_bf16(ap, bv2, O[2].v);
    O[3].v = wmma_bf16(ap, bv3, O[3].v);
    __syncthreads();
  }
  // normalize + write combined-head row [b, s, h*HD + hd] as bf16
#pragma unroll
  for (int r = 0; r < 8; ++r) {
    float inv = 1.0f / lrow[r];
    int row = q0 + r + 8 * hi;
    size_t o = ((size_t)b * S_LEN + row) * D_MODEL + h * HD;
#pragma unroll
    for (int j = 0; j < 4; ++j)
      Ab[o + j * 16 + ln] = f32_to_bf16(O[j].f[r] * inv);
  }
}

// ---------------- output projection: out = Ab · Woᵀ (f32 out) ----------------
__global__ void __launch_bounds__(256) out_proj(
    const unsigned short* __restrict__ Ab,
    const unsigned short* __restrict__ Wo,
    float* __restrict__ out) {
  int wave  = (blockIdx.x * blockDim.x + threadIdx.x) >> 5;
  int nquad = wave & 15;
  int mtile = wave >> 4;                 // 0..127, 32 rows each
  int m0 = mtile * 32, n0 = nquad * 64;

  Acc acc[2][4];
#pragma unroll
  for (int i = 0; i < 2; ++i)
#pragma unroll
    for (int j = 0; j < 4; ++j)
#pragma unroll
      for (int r = 0; r < 8; ++r) acc[i][j].f[r] = 0.0f;

  v16bf a0 = load_a(Ab, D_MODEL, m0, 0);
  v16bf a1 = load_a(Ab, D_MODEL, m0 + 16, 0);
  for (int k0 = 0; k0 < D_MODEL; k0 += 32) {
    v16bf b0 = load_b(Wo, D_MODEL, n0,      k0);
    v16bf b1 = load_b(Wo, D_MODEL, n0 + 16, k0);
    v16bf b2 = load_b(Wo, D_MODEL, n0 + 32, k0);
    v16bf b3 = load_b(Wo, D_MODEL, n0 + 48, k0);
    int kn = (k0 + 32 < D_MODEL) ? k0 + 32 : 0;
    v16bf a0n = load_a(Ab, D_MODEL, m0, kn);
    v16bf a1n = load_a(Ab, D_MODEL, m0 + 16, kn);
    acc[0][0].v = wmma_bf16(a0, b0, acc[0][0].v);
    acc[1][0].v = wmma_bf16(a1, b0, acc[1][0].v);
    acc[0][1].v = wmma_bf16(a0, b1, acc[0][1].v);
    acc[1][1].v = wmma_bf16(a1, b1, acc[1][1].v);
    acc[0][2].v = wmma_bf16(a0, b2, acc[0][2].v);
    acc[1][2].v = wmma_bf16(a1, b2, acc[1][2].v);
    acc[0][3].v = wmma_bf16(a0, b3, acc[0][3].v);
    acc[1][3].v = wmma_bf16(a1, b3, acc[1][3].v);
    a0 = a0n; a1 = a1n;
  }
  int lane = threadIdx.x & 31, hi = lane >> 4, ln = lane & 15;
#pragma unroll
  for (int i = 0; i < 2; ++i)
#pragma unroll
    for (int j = 0; j < 4; ++j)
#pragma unroll
      for (int r = 0; r < 8; ++r)
        out[(size_t)(m0 + 16 * i + r + 8 * hi) * D_MODEL + n0 + 16 * j + ln] = acc[i][j].f[r];
}

extern "C" void kernel_launch(void* const* d_in, const int* in_sizes, int n_in,
                              void* d_out, int out_size, void* d_ws, size_t ws_size,
                              hipStream_t stream) {
  const float* X    = (const float*)d_in[0];
  const int*   mask = (const int*)d_in[1];
  const float* Wq   = (const float*)d_in[2];
  const float* Wk   = (const float*)d_in[3];
  const float* Wv   = (const float*)d_in[4];
  const float* Wo   = (const float*)d_in[5];
  float* out = (float*)d_out;

  const size_t NX = (size_t)B_SZ * S_LEN * D_MODEL;   // 4,194,304
  const size_t NW = (size_t)D_MODEL * D_MODEL;        // 1,048,576

  char* ws = (char*)d_ws;
  size_t off = 0;
  auto alloc = [&](size_t bytes) {
    void* p = ws + off;
    off = (off + bytes + 255) & ~(size_t)255;
    return p;
  };
  unsigned short* Xb  = (unsigned short*)alloc(NX * 2);
  unsigned short* Wqb = (unsigned short*)alloc(NW * 2);
  unsigned short* Wkb = (unsigned short*)alloc(NW * 2);
  unsigned short* Wvb = (unsigned short*)alloc(NW * 2);
  unsigned short* Wob = (unsigned short*)alloc(NW * 2);
  unsigned short* Qb  = (unsigned short*)alloc(NX * 2);
  unsigned short* Kb  = (unsigned short*)alloc(NX * 2);
  unsigned short* Vt  = (unsigned short*)alloc(NX * 2);
  unsigned short* Ab  = (unsigned short*)alloc(NX * 2);

  const int thr = 256;
  cvt_bf16<<<(int)((NX / 8 + thr - 1) / thr), thr, 0, stream>>>(X,  Xb,  (int)(NX / 8));
  cvt_bf16<<<(int)((NW / 8 + thr - 1) / thr), thr, 0, stream>>>(Wq, Wqb, (int)(NW / 8));
  cvt_bf16<<<(int)((NW / 8 + thr - 1) / thr), thr, 0, stream>>>(Wk, Wkb, (int)(NW / 8));
  cvt_bf16<<<(int)((NW / 8 + thr - 1) / thr), thr, 0, stream>>>(Wv, Wvb, (int)(NW / 8));
  cvt_bf16<<<(int)((NW / 8 + thr - 1) / thr), thr, 0, stream>>>(Wo, Wob, (int)(NW / 8));

  // 3 mats * 128 m-tiles * 16 n-quads = 6144 waves -> 768 blocks of 256 thr
  qkv_proj<<<768, 256, 0, stream>>>(Xb, Wqb, Wkb, Wvb, Qb, Kb, Vt);

  // 128 q-tiles x 32 (b,h) pairs, one wave each
  attn<<<dim3(S_LEN / 16, B_SZ * H_CNT), 32, 0, stream>>>(Qb, Kb, Vt, mask, Ab);

  // 128 m-tiles * 16 n-quads = 2048 waves -> 256 blocks of 256 thr
  out_proj<<<256, 256, 0, stream>>>(Ab, Wob, out);
}
